// EfficientMemoryGEMMWithSoftmax_22720376996352
// MI455X (gfx1250) — matrix operand order repaired and checked by reading, not compiled, not run on hardware
//
#include <hip/hip_runtime.h>
#include <hip/hip_bf16.h>

typedef __attribute__((ext_vector_type(16))) _Float16 v16h;
typedef __attribute__((ext_vector_type(8)))  _Float16 v8h;
typedef __attribute__((ext_vector_type(8)))  float    v8f;

#define S_DIM 1024
#define D_DIM 128
#define BH_DIM 128   // B*H = 8*16
#define TM 128       // output rows per block
#define TK 32        // K per stage (== WMMA K)
#define LDA 40       // LDS row stride in halves (TK + 8 pad; 80B, multiple of 16B)
#define NSTAGE (S_DIM / TK)

__global__ __launch_bounds__(256)
void gemm_softmax_pv_wmma(const float* __restrict__ x1,
                          const float* __restrict__ x2,
                          float* __restrict__ out) {
    // A tile: [m][k] (f16), B tile stored transposed: [n][k] (f16)
    __shared__ __align__(16) _Float16 Ah [TM    * LDA];
    __shared__ __align__(16) _Float16 BhT[D_DIM * LDA];

    const int tid   = threadIdx.x;
    const int lane  = tid & 31;
    const int wave  = tid >> 5;
    const int waveM = wave & 3;   // 4 waves along M: 32 rows each
    const int waveN = wave >> 2;  // 2 waves along N: 64 cols each

    const int mt0 = blockIdx.x * TM;
    const int bh  = blockIdx.y;

    const float* __restrict__ Abase = x1 + ((size_t)bh * S_DIM + mt0) * S_DIM;
    const float* __restrict__ Bbase = x2 + (size_t)bh * S_DIM * D_DIM;
    float* __restrict__ Cbase       = out + ((size_t)bh * S_DIM + mt0) * D_DIM;

    // A loads: 128 rows x 8 float4 (contiguous K-segments)
    const int a_r = tid >> 3;            // reused every stage (ia = tid + p*256)
    // B loads: each thread owns 4 column/K-segment pairs: seg = tid + p*256,
    //   c = seg & 127 (column), kg = seg >> 7 (K-group of 4)
    // -> 4 strided b32 global loads per segment, fully coalesced across lanes,
    //    then a single contiguous 4-half (8B) LDS store into the transposed tile.

    // fp32 accumulators: 2 (M sub-tiles) x 4 (N sub-tiles) of 16x16
    v8f acc[2][4];
    #pragma unroll
    for (int i = 0; i < 2; ++i)
        #pragma unroll
        for (int j = 0; j < 4; ++j)
            #pragma unroll
            for (int v = 0; v < 8; ++v)
                acc[i][j][v] = 0.0f;

    // ---- software pipeline: prefetch stage 0 into registers ----
    float4 aReg[4];
    float  bReg[4][4];
    #pragma unroll
    for (int p = 0; p < 4; ++p) {
        const int ia = tid + p * 256;
        const int r  = ia >> 3, ca = (ia & 7) << 2;
        aReg[p] = *(const float4*)(Abase + (size_t)r * S_DIM + ca);

        const int seg = tid + p * 256;
        const int c   = seg & 127, kg = (seg >> 7) << 2;
        #pragma unroll
        for (int q = 0; q < 4; ++q)
            bReg[p][q] = Bbase[(size_t)(kg + q) * D_DIM + c];
    }

    for (int kt = 0; kt < NSTAGE; ++kt) {
        // ---- registers -> LDS (fp32 -> f16 convert) ----
        #pragma unroll
        for (int p = 0; p < 4; ++p) {
            const int ia = tid + p * 256;
            const int r  = ia >> 3, ca = (ia & 7) << 2;
            _Float16* ad = &Ah[r * LDA + ca];
            ad[0] = (_Float16)aReg[p].x; ad[1] = (_Float16)aReg[p].y;
            ad[2] = (_Float16)aReg[p].z; ad[3] = (_Float16)aReg[p].w;

            const int seg = tid + p * 256;
            const int c   = seg & 127, kg = (seg >> 7) << 2;
            _Float16* bd = &BhT[c * LDA + kg];
            bd[0] = (_Float16)bReg[p][0]; bd[1] = (_Float16)bReg[p][1];
            bd[2] = (_Float16)bReg[p][2]; bd[3] = (_Float16)bReg[p][3];
        }
        __syncthreads();

        // ---- prefetch next stage from global while this stage computes ----
        if (kt + 1 < NSTAGE) {
            const int k0 = (kt + 1) * TK;
            #pragma unroll
            for (int p = 0; p < 4; ++p) {
                const int ia = tid + p * 256;
                const int r  = ia >> 3, ca = (ia & 7) << 2;
                aReg[p] = *(const float4*)(Abase + (size_t)r * S_DIM + k0 + ca);

                const int seg = tid + p * 256;
                const int c   = seg & 127, kg = (seg >> 7) << 2;
                #pragma unroll
                for (int q = 0; q < 4; ++q)
                    bReg[p][q] = Bbase[(size_t)(k0 + kg + q) * D_DIM + c];
            }
        }

        // ---- build fragments, contiguous k = kh*16 + s in A and B ----
        const int mrn = lane & 15;   // M (for A) / N (for B) within tile
        const int kh  = lane >> 4;   // K half selector

        v16h afrag[2];
        #pragma unroll
        for (int i = 0; i < 2; ++i) {
            const _Float16* ap = &Ah[(waveM * 32 + i * 16 + mrn) * LDA];
            v8h lo = *(const v8h*)(ap + kh * 16);       // K = kh*16 .. +7
            v8h hi = *(const v8h*)(ap + kh * 16 + 8);   // K = kh*16+8 .. +15
            afrag[i] = __builtin_shufflevector(lo, hi,
                0, 1, 2, 3, 4, 5, 6, 7, 8, 9, 10, 11, 12, 13, 14, 15);
        }

        v16h bfrag[4];
        #pragma unroll
        for (int j = 0; j < 4; ++j) {
            const _Float16* bp = &BhT[(waveN * 64 + j * 16 + mrn) * LDA + kh * 16];
            v8h lo = *(const v8h*)(bp);                 // K = kh*16 .. +7
            v8h hi = *(const v8h*)(bp + 8);             // K = kh*16+8 .. +15
            bfrag[j] = __builtin_shufflevector(lo, hi,
                0, 1, 2, 3, 4, 5, 6, 7, 8, 9, 10, 11, 12, 13, 14, 15);
        }

        // ---- 8 WMMAs: D = A*B + C, fp32 accumulate ----
        #pragma unroll
        for (int i = 0; i < 2; ++i)
            #pragma unroll
            for (int j = 0; j < 4; ++j)
                acc[i][j] = __builtin_amdgcn_wmma_f32_16x16x32_f16(
                    /*neg_a=*/false, afrag[i],
                    /*neg_b=*/false, bfrag[j],
                    /*c_mod=*/(short)0, acc[i][j],
                    /*reuse_a=*/false, /*reuse_b=*/false);

        __syncthreads();
    }

    // ---- epilogue: C/D layout -> global fp32 (64B coalesced per half-wave) ----
    #pragma unroll
    for (int i = 0; i < 2; ++i) {
        #pragma unroll
        for (int j = 0; j < 4; ++j) {
            const int n = waveN * 64 + j * 16 + (lane & 15);
            #pragma unroll
            for (int v = 0; v < 8; ++v) {
                const int m = waveM * 32 + i * 16 + v + 8 * (lane >> 4);
                Cbase[(size_t)m * D_DIM + n] = acc[i][j][v];
            }
        }
    }
    (void)a_r;
}

extern "C" void kernel_launch(void* const* d_in, const int* in_sizes, int n_in,
                              void* d_out, int out_size, void* d_ws, size_t ws_size,
                              hipStream_t stream) {
    const float* x1 = (const float*)d_in[0];   // [B,H,S,S] softmax probs
    const float* x2 = (const float*)d_in[1];   // [B,H,S,D] values
    float* out      = (float*)d_out;           // [B,H,S,D]

    dim3 grid(S_DIM / TM, BH_DIM);
    dim3 block(256);
    gemm_softmax_pv_wmma<<<grid, block, 0, stream>>>(x1, x2, out);
}
